// HSLSTMForDirection_80702435492487
// MI455X (gfx1250) — matrix operand order repaired and checked by reading, not compiled, not run on hardware
//
#include <hip/hip_runtime.h>
#include <hip/hip_bf16.h>

// ---------------------------------------------------------------------------
// Problem constants (match reference)
// ---------------------------------------------------------------------------
#define BB   256            // batch
#define TT   512            // time steps
#define II   256            // input size
#define HH   512            // hidden size
#define GG   (4 * HH)       // gate width = 2048
#define OO   3              // output classes

typedef __attribute__((ext_vector_type(16))) __bf16        v16bf;
typedef __attribute__((ext_vector_type(8)))  float         v8f;
typedef __attribute__((ext_vector_type(4)))  unsigned int  u32x4;  // plain
// builtin vector (not HIP's class-based uint4) so it can be loaded through an
// address-space(1) qualified pointer on both host and device passes.

// explicit global-address-space views: guarantees global_load_b128 /
// global_store (LOADcnt/STOREcnt only) instead of flat_* (which also ties up
// DScnt and the LDS path on CDNA5).
#define GAS __attribute__((address_space(1)))

static __device__ __forceinline__ u32x4 load_g128(const void* p) {
    return *(const GAS u32x4*)(uintptr_t)p;
}
static __device__ __forceinline__ float load_gf32(const void* p) {
    return *(const GAS float*)(uintptr_t)p;
}
static __device__ __forceinline__ void store_gf32(void* p, float v) {
    *(GAS float*)(uintptr_t)p = v;
}

// float -> bf16 (round to nearest even), stored as ushort so we never rely on
// __bf16 arithmetic, only on the WMMA operand type.
static __device__ __forceinline__ unsigned short f2bf(float f) {
    unsigned int u = __float_as_uint(f);
    unsigned int r = 0x7FFFu + ((u >> 16) & 1u);
    return (unsigned short)((u + r) >> 16);
}

static __device__ __forceinline__ float fsigmoid(float x) {
    return 1.0f / (1.0f + __expf(-x));
}
static __device__ __forceinline__ float ftanh(float x) {
    x = fminf(15.0f, fmaxf(-15.0f, x));
    float e = __expf(2.0f * x);
    return (e - 1.0f) / (e + 1.0f);
}

// ---------------------------------------------------------------------------
// Prep kernels: fp32 -> bf16 weights, bias fusion (bih + bhh)
// ---------------------------------------------------------------------------
__global__ void cvt_bf16_kernel(const float* __restrict__ src,
                                unsigned short* __restrict__ dst, int n) {
    int i = blockIdx.x * blockDim.x + threadIdx.x;
    if (i < n) dst[i] = f2bf(src[i]);
}

__global__ void bias_sum_kernel(const float* __restrict__ a,
                                const float* __restrict__ b,
                                float* __restrict__ dst, int n) {
    int i = blockIdx.x * blockDim.x + threadIdx.x;
    if (i < n) dst[i] = a[i] + b[i];
}

// ---------------------------------------------------------------------------
// One wave's GEMM micro-kernel (K is a compile-time constant so the 8 weight
// tile addresses fold into base + immediate offsets of global_load_b128):
//   acc[8] (+=) A(16 x K, bf16, LDS row-major stride K)  x  B(K x 128)
// B columns are gate columns 16*(w+16r)+n from weight matrix W (row-major
// [2048 x K] bf16): B[k,n] = W[16*(w+16r)+n, k].
// A-fragment (16-bit A 16x32, contiguous K order used here): lanes 0-15 hold K 0-15, lanes
// 16-31 hold K 16-31, row = lane%16.
// B-fragment (16-bit B 32x16, same contiguous K order as A): lane%16 = column, lanes<16 hold K 0-15,
// lanes>=16 hold K 16-31 (contiguous along K in memory).
// ---------------------------------------------------------------------------
template <int K>
static __device__ __forceinline__ void gemm_tile(
    v8f acc[8],
    const unsigned short* __restrict__ a_lds,
    const unsigned short* __restrict__ Wg,
    int w, int lane)
{
    const int lo = lane & 15;
    const int hi = lane >> 4;
    // per-lane base: column (16w + lo), K-offset (16*hi); per-tile r adds the
    // compile-time constant 256*r*K elements -> immediate offset in the load.
    const unsigned short* base = Wg + (size_t)(16 * w + lo) * K + 16 * hi;
    for (int k0 = 0; k0 < K; k0 += 32) {
        union { u32x4 u[2]; v16bf v; } a;
        const u32x4* ap0 = (const u32x4*)(a_lds + lo * K + k0 + 16 * hi);
        const u32x4* ap1 = (const u32x4*)(a_lds + lo * K + k0 + 16 * hi + 8);
        a.u[0] = *ap0;
        a.u[1] = *ap1;
#pragma unroll
        for (int r = 0; r < 8; ++r) {
            const unsigned short* bp0 = base + (size_t)r * 256 * K + k0;
            union { u32x4 u[2]; v16bf v; } b;
            b.u[0] = load_g128(bp0);
            b.u[1] = load_g128(bp0 + 8);
            acc[r] = __builtin_amdgcn_wmma_f32_16x16x32_bf16(
                false, a.v, false, b.v, (short)0, acc[r], false, false);
        }
    }
}

// ---------------------------------------------------------------------------
// Persistent 3-layer LSTM recurrence. One block = one 16-row batch tile,
// 512 threads = 16 waves; wave w owns gate N-tiles {w+16r : r=0..7} which
// cover hidden columns {16w+256p + 0..15 : p=0,1} for all 4 gates.
// h-state (bf16), c-state (fp32) and fused biases all live in LDS so that
// only the 64 accumulator VGPRs are persistently live -> no spilling, and
// the compiler keeps weight B-tiles in flight in registers.
// ---------------------------------------------------------------------------
__global__ void __launch_bounds__(512, 1)
lstm3_persistent_kernel(const float* __restrict__ x,              // [B,T,I]
                        const unsigned short* __restrict__ wih0,  // [G,I]
                        const unsigned short* __restrict__ whh0,  // [G,H]
                        const unsigned short* __restrict__ wih1,  // [G,H]
                        const unsigned short* __restrict__ whh1,  // [G,H]
                        const unsigned short* __restrict__ wih2,  // [G,H]
                        const unsigned short* __restrict__ whh2,  // [G,H]
                        const float* __restrict__ biasAll,        // [3,G]
                        float* __restrict__ h2_out)               // [B,H]
{
    __shared__ __align__(16) unsigned short xbuf[16 * II];        //   8 KB
    __shared__ __align__(16) unsigned short hbuf[3 * 16 * HH];    //  48 KB
    __shared__ __align__(16) float          cbuf[3 * 16 * HH];    //  96 KB
    __shared__ __align__(16) float          bbuf[3 * GG];         //  24 KB

    const int tid  = threadIdx.x;
    const int w    = tid >> 5;        // wave id 0..15
    const int lane = tid & 31;
    const int lo   = lane & 15;
    const int hi   = lane >> 4;
    const int b0   = blockIdx.x * 16; // batch tile base row

    const unsigned short* Wih[3] = { wih0, wih1, wih2 };
    const unsigned short* Whh[3] = { whh0, whh1, whh2 };

    // zero h and c state, stage fused biases into LDS
    for (int e = tid; e < 3 * 16 * HH; e += 512) {
        hbuf[e] = 0;
        cbuf[e] = 0.0f;
    }
    for (int e = tid; e < 3 * GG; e += 512) bbuf[e] = load_gf32(biasAll + e);
    __syncthreads();

    for (int t = 0; t < TT; ++t) {
        // stage x_t tile into LDS as bf16 (16 x 256)
        for (int e = tid; e < 16 * II; e += 512) {
            const int row = e >> 8;          // /256
            const int col = e & (II - 1);
            xbuf[e] = f2bf(load_gf32(
                x + (size_t)(b0 + row) * (TT * II) + (size_t)t * II + col));
        }
        __syncthreads();

        for (int l = 0; l < 3; ++l) {
            // accumulators initialized with fused bias (splat: D row-layout
            // means all 8 acc VGPRs of a tile share the same gate column set)
            v8f acc[8];
#pragma unroll
            for (int r = 0; r < 8; ++r) {
                const float bv = bbuf[l * GG + 16 * (w + 16 * r) + lo];
                v8f a = { bv, bv, bv, bv, bv, bv, bv, bv };
                acc[r] = a;
            }

            // input projection: layer0 from x (K=256), else previous layer h
            if (l == 0)
                gemm_tile<II>(acc, xbuf, Wih[0], w, lane);
            else
                gemm_tile<HH>(acc, hbuf + (l - 1) * 16 * HH, Wih[l], w, lane);

            // recurrent projection (K=512)
            gemm_tile<HH>(acc, hbuf + l * 16 * HH, Whh[l], w, lane);

            __syncthreads();   // all waves finished reading hbuf[l] / input

            // fused LSTM nonlinearity: tiles (p, 2+p, 4+p, 6+p) = (i,f,g,o)
            // at hidden column 16w + 256p + lo, rows v + 8*hi
            float* cl = cbuf + l * 16 * HH;
#pragma unroll
            for (int p = 0; p < 2; ++p) {
                const int colp = 16 * w + 256 * p + lo;
#pragma unroll
                for (int v = 0; v < 8; ++v) {
                    const int   row = v + 8 * hi;
                    const int   idx = row * HH + colp;
                    const float iv  = fsigmoid(acc[0 + p][v]);
                    const float fv  = fsigmoid(acc[2 + p][v]);
                    const float gv  = ftanh(acc[4 + p][v]);
                    const float ov  = fsigmoid(acc[6 + p][v]);
                    const float c   = fv * cl[idx] + iv * gv;
                    cl[idx]         = c;
                    const float h   = ov * ftanh(c);
                    hbuf[l * 16 * HH + idx] = f2bf(h);
                    if (l == 2 && t == TT - 1)
                        store_gf32(h2_out + (size_t)(b0 + row) * HH + colp, h);
                }
            }
            __syncthreads();   // h_new visible to next layer / next step
        }
    }
}

// ---------------------------------------------------------------------------
// Final FC: out[b,o] = h2[b,:] . fcW[o,:] + fcB[o]   (256x512x3, trivial)
// ---------------------------------------------------------------------------
__global__ void fc_kernel(const float* __restrict__ h2,
                          const float* __restrict__ fcW,
                          const float* __restrict__ fcB,
                          float* __restrict__ out) {
    int idx = blockIdx.x * blockDim.x + threadIdx.x;
    if (idx >= BB * OO) return;
    const int b = idx / OO;
    const int o = idx % OO;
    float s = fcB[o];
    for (int k = 0; k < HH; ++k)
        s += h2[(size_t)b * HH + k] * fcW[(size_t)o * HH + k];
    out[idx] = s;
}

// ---------------------------------------------------------------------------
// Launch
// ---------------------------------------------------------------------------
extern "C" void kernel_launch(void* const* d_in, const int* in_sizes, int n_in,
                              void* d_out, int out_size, void* d_ws, size_t ws_size,
                              hipStream_t stream) {
    (void)in_sizes; (void)n_in; (void)out_size; (void)ws_size;

    const float* x    = (const float*)d_in[0];
    const float* Wih0 = (const float*)d_in[1];
    const float* Whh0 = (const float*)d_in[2];
    const float* bih0 = (const float*)d_in[3];
    const float* bhh0 = (const float*)d_in[4];
    const float* Wih1 = (const float*)d_in[5];
    const float* Whh1 = (const float*)d_in[6];
    const float* bih1 = (const float*)d_in[7];
    const float* bhh1 = (const float*)d_in[8];
    const float* Wih2 = (const float*)d_in[9];
    const float* Whh2 = (const float*)d_in[10];
    const float* bih2 = (const float*)d_in[11];
    const float* bhh2 = (const float*)d_in[12];
    const float* fcW  = (const float*)d_in[13];
    const float* fcB  = (const float*)d_in[14];

    // workspace carve-out (256B aligned)
    char* ws = (char*)d_ws;
    size_t off = 0;
    auto carve = [&](size_t bytes) -> char* {
        char* p = ws + off;
        off = (off + bytes + 255) & ~(size_t)255;
        return p;
    };
    unsigned short* wih0b = (unsigned short*)carve((size_t)GG * II * 2);
    unsigned short* whh0b = (unsigned short*)carve((size_t)GG * HH * 2);
    unsigned short* wih1b = (unsigned short*)carve((size_t)GG * HH * 2);
    unsigned short* whh1b = (unsigned short*)carve((size_t)GG * HH * 2);
    unsigned short* wih2b = (unsigned short*)carve((size_t)GG * HH * 2);
    unsigned short* whh2b = (unsigned short*)carve((size_t)GG * HH * 2);
    float* biasAll        = (float*)carve((size_t)3 * GG * 4);
    float* h2buf          = (float*)carve((size_t)BB * HH * 4);

    const int TPB = 256;
    auto cvt = [&](const float* s, unsigned short* d, int n) {
        cvt_bf16_kernel<<<(n + TPB - 1) / TPB, TPB, 0, stream>>>(s, d, n);
    };
    cvt(Wih0, wih0b, GG * II);
    cvt(Whh0, whh0b, GG * HH);
    cvt(Wih1, wih1b, GG * HH);
    cvt(Whh1, whh1b, GG * HH);
    cvt(Wih2, wih2b, GG * HH);
    cvt(Whh2, whh2b, GG * HH);

    bias_sum_kernel<<<(GG + TPB - 1) / TPB, TPB, 0, stream>>>(bih0, bhh0, biasAll,          GG);
    bias_sum_kernel<<<(GG + TPB - 1) / TPB, TPB, 0, stream>>>(bih1, bhh1, biasAll + GG,     GG);
    bias_sum_kernel<<<(GG + TPB - 1) / TPB, TPB, 0, stream>>>(bih2, bhh2, biasAll + 2 * GG, GG);

    // persistent recurrence: 16 batch tiles x 16 waves
    lstm3_persistent_kernel<<<BB / 16, 512, 0, stream>>>(
        x, wih0b, whh0b, wih1b, whh1b, wih2b, whh2b, biasAll, h2buf);

    fc_kernel<<<(BB * OO + TPB - 1) / TPB, TPB, 0, stream>>>(
        h2buf, fcW, fcB, (float*)d_out);
}
